// BinaryConv2D_55765855371572
// MI455X (gfx1250) — 8-bit matrix operand order repaired and checked by reading, compiled, not run on hardware
//
#include <hip/hip_runtime.h>


typedef __attribute__((ext_vector_type(8))) int v8i;
typedef __attribute__((ext_vector_type(4))) int v4i;

#define N_IMG 64
#define HW    128
#define CH    64
#define NF    64
#define ROWS  8
#define LROWS (ROWS + 2)
#define LCOLS (HW + 2)
#define PIXB  72
#define XROWB (LCOLS * PIXB)
#define XS_BYTES (LROWS * XROWB)
#define WFB   80
#define WTAPB (NF * WFB)
#define WS_BYTES (9 * WTAPB)
#define LDS_BYTES (XS_BYTES + WS_BYTES + NF * 4)

__global__ void prep_weights(const float* __restrict__ w,
                             signed char* __restrict__ wsign,
                             float* __restrict__ alpha) {
  __shared__ float part[256];
  const int tid = threadIdx.x;
  for (int i = tid; i < 9 * NF * CH; i += 256) {
    const int t  = i >> 12;
    const int f  = (i >> 6) & 63;
    const int ci = i & 63;
    const float v = w[(t * CH + ci) * NF + f];
    wsign[i] = (v >= 0.0f) ? (signed char)1 : (signed char)-1;
  }
  const int f = tid & 63, seg = tid >> 6;
  float s = 0.0f;
  for (int i = seg * 144; i < (seg + 1) * 144; ++i)
    s += fabsf(w[i * NF + f]);
  part[tid] = s;
  __syncthreads();
  if (tid < NF)
    alpha[tid] = (part[tid] + part[tid + 64] + part[tid + 128] + part[tid + 192]) *
                 (1.0f / 576.0f);
}

__launch_bounds__(256)
__global__ void binconv_wmma(const float* __restrict__ x,
                             const signed char* __restrict__ wsign,
                             const float* __restrict__ alpha,
                             float* __restrict__ out) {
  extern __shared__ char lds[];
  char*  xs = lds;
  char*  ws = lds + XS_BYTES;
  float* al = (float*)(lds + XS_BYTES + WS_BYTES);

  const int tid = threadIdx.x;
  const int wg  = blockIdx.x;
  const int n   = wg >> 4;
  const int h0  = (wg & 15) * ROWS;

  for (int g = tid; g < 9 * NF * 8; g += 256) {
    const int t  = g >> 9;
    const int f  = (g >> 3) & 63;
    const int kb = g & 7;
    const int p  = ((kb & 2) ? 8 : 0) | ((kb & 1) ? 16 : 0) | ((kb & 4) ? 32 : 0);
    *(long*)(ws + t * WTAPB + f * WFB + p) = *(const long*)(wsign + (long)g * 8);
  }
  if (tid < NF) al[tid] = alpha[tid];

  if (tid < 80) {
    const int chunk = tid & 3, pixel = tid >> 2;
    const int r = pixel >> 1, cc = (pixel & 1) ? (LCOLS - 1) : 0;
    *(v4i*)(xs + r * XROWB + cc * PIXB + chunk * 16) = (v4i){0, 0, 0, 0};
  }

  const long imgBase = (long)n * HW * HW * CH;
#pragma unroll
  for (int j = 0; j < 8; ++j) {
    const int g   = tid + j * 256;
    const int gw  = g >> 4;
    const int gi  = g & 15;
    const int gp  = (gi & 9) | ((gi & 4) >> 1) | ((gi & 2) << 1);
    const float* gpix = x + imgBase + ((long)(h0 - 1) * HW + gw) * CH + (gi << 2);
    char* lp = xs + (gw + 1) * PIXB + gp * 4;
#pragma unroll
    for (int r = 0; r < LROWS; ++r) {
      const int gh = h0 - 1 + r;
      unsigned packed = 0u;
      if (gh >= 0 && gh < HW) {
        const float4 v = *(const float4*)gpix;
        const unsigned b0 = 1u + ((unsigned)((int)__float_as_uint(v.x) >> 31) & 0xFEu);
        const unsigned b1 = 1u + ((unsigned)((int)__float_as_uint(v.y) >> 31) & 0xFEu);
        const unsigned b2 = 1u + ((unsigned)((int)__float_as_uint(v.z) >> 31) & 0xFEu);
        const unsigned b3 = 1u + ((unsigned)((int)__float_as_uint(v.w) >> 31) & 0xFEu);
        packed = b0 | (b1 << 8) | (b2 << 16) | (b3 << 24);
      }
      *(unsigned*)lp = packed;
      gpix += (long)HW * CH;
      lp   += XROWB;
    }
  }
  __syncthreads();

  const int wave = tid >> 5;
  const int lane = tid & 31;
  const int lh   = lane >> 4;
  const int ln   = lane & 15;

  float af[4];
#pragma unroll
  for (int nt = 0; nt < 4; ++nt) af[nt] = al[nt * 16 + ln];

  const int r = wave;
  for (int mt2 = 0; mt2 < 4; ++mt2) {
    const int c0 = mt2 * 32;
    v8i acc[2][4] = {};

#pragma unroll
    for (int tap = 0; tap < 9; ++tap) {
      const int dr = tap / 3;
      const int dc = tap - dr * 3;
      const char* ap0 = xs + (r + dr) * XROWB + (c0 + ln + dc) * PIXB + lh * 16;
      const char* ap1 = ap0 + 16 * PIXB;
      union { v8i v; long l[4]; } A0, A1;
      A0.l[0] = *(const long*)(ap0 + 0);
      A0.l[1] = *(const long*)(ap0 + 8);
      A0.l[2] = *(const long*)(ap0 + 32);
      A0.l[3] = *(const long*)(ap0 + 40);
      A1.l[0] = *(const long*)(ap1 + 0);
      A1.l[1] = *(const long*)(ap1 + 8);
      A1.l[2] = *(const long*)(ap1 + 32);
      A1.l[3] = *(const long*)(ap1 + 40);

#pragma unroll
      for (int nt = 0; nt < 4; ++nt) {
        const char* bp = ws + tap * WTAPB + (nt * 16 + ln) * WFB + lh * 16;
        union { v8i v; long l[4]; } B;
        B.l[0] = *(const long*)(bp + 0);
        B.l[1] = *(const long*)(bp + 8);
        B.l[2] = *(const long*)(bp + 32);
        B.l[3] = *(const long*)(bp + 40);
        acc[0][nt] = __builtin_amdgcn_wmma_i32_16x16x64_iu8(
            true, A0.v, true, B.v, acc[0][nt], false, false);
        acc[1][nt] = __builtin_amdgcn_wmma_i32_16x16x64_iu8(
            true, A1.v, true, B.v, acc[1][nt], false, false);
      }
    }

#pragma unroll
    for (int sub = 0; sub < 2; ++sub) {
      const long rowBase =
          (((long)n * HW + (h0 + r)) * HW + c0 + sub * 16 + lh * 8) * NF;
#pragma unroll
      for (int nt = 0; nt < 4; ++nt) {
        float* op = out + rowBase + nt * 16 + ln;
#pragma unroll
        for (int q = 0; q < 8; ++q) {
          __builtin_nontemporal_store(af[nt] * (float)acc[sub][nt][q],
                                      op + (long)q * NF);
        }
      }
    }
  }
}

extern "C" void kernel_launch(void* const* d_in, const int* in_sizes, int n_in,
                              void* d_out, int out_size, void* d_ws, size_t ws_size,
                              hipStream_t stream) {
  const float* x = (const float*)d_in[0];
  const float* w = (const float*)d_in[1];

  signed char* wsign = (signed char*)d_ws;
  float*       alpha = (float*)((char*)d_ws + 9 * NF * CH);

  prep_weights<<<1, 256, 0, stream>>>(w, wsign, alpha);

  const int grid = N_IMG * (HW / ROWS);
  binconv_wmma<<<grid, 256, LDS_BYTES, stream>>>(x, wsign, alpha, (float*)d_out);
}
